// TrajsEncoder_59279138619519
// MI455X (gfx1250) — matrix operand order repaired and checked by reading, not compiled, not run on hardware
//
#include <hip/hip_runtime.h>

// ---------------------------------------------------------------------------
// TrajsEncoder for MI455X (gfx1250).
// - All GEMMs: bf16 WMMA (v_wmma_f32_16x16x32_bf16). One wave computes a
//   16-row x (NT*32)-col output strip, NT a template constant (1/2/4): the
//   inner K loop is branch-free straight-line code -- one A fragment load
//   (4x b128 + packed cvt), 2*NT clause-able B fragment loads, 2*NT WMMAs.
//   Big activation matrices stream from HBM ~once per layer; packed bf16
//   weights are L2-resident.
// - Training-mode BatchNorm: column sum/sumsq fused into the GEMM epilogue.
// - Graph ops: coalesced gather, f32 atomic scatter mean/max, segment softmax.
// ---------------------------------------------------------------------------

#define NNODES 200000LL
#define NEDGES 400000LL
#define NGRAPH 2000LL
#define NCH    64
#define LAT    128
#define BN_EPS 1e-5f

typedef __attribute__((ext_vector_type(16))) __bf16 bf16x16;
typedef __attribute__((ext_vector_type(8)))  float  floatx8;
typedef __attribute__((ext_vector_type(4)))  float  f32x4;

// ------------------------------- device utils ------------------------------

__device__ __forceinline__ long long gtid() { return (long long)blockIdx.x * blockDim.x + threadIdx.x; }
__device__ __forceinline__ long long gstr() { return (long long)gridDim.x * blockDim.x; }

__device__ __forceinline__ void atomicMaxF(float* addr, float v) {
  if (v >= 0.0f) atomicMax((int*)addr, __float_as_int(v));
  else           atomicMin((unsigned int*)addr, (unsigned int)__float_as_int(v));
}

// ------------------------------- elementwise -------------------------------

__global__ void k_fill(float* __restrict__ p, long long n, float v) {
  for (long long i = gtid(); i < n; i += gstr()) p[i] = v;
}

__global__ void k_count_edges(const int* __restrict__ dst, long long E, float* __restrict__ cnt) {
  for (long long i = gtid(); i < E; i += gstr()) atomicAdd(&cnt[dst[i]], 1.0f);
}

// y[m, 0:K] = x[m,:], y[m, K:2K] = x^2, y[m, 2K:ld] = 0   (x dense width K)
__global__ void k_concat_sq(const float* __restrict__ x, long long M, int K, int ld,
                            float* __restrict__ y) {
  long long total = M * ld;
  for (long long i = gtid(); i < total; i += gstr()) {
    long long m = i / ld; int c = (int)(i % ld);
    float v = 0.f;
    if (c < K)          { float t = x[m * K + c];       v = t; }
    else if (c < 2 * K) { float t = x[m * K + (c - K)]; v = t * t; }
    y[i] = v;
  }
}

__global__ void k_concat2(const float* __restrict__ a, int ca, const float* __restrict__ b, int cb,
                          long long M, float* __restrict__ y) {
  int cy = ca + cb;
  long long total = M * cy;
  for (long long i = gtid(); i < total; i += gstr()) {
    long long m = i / cy; int c = (int)(i % cy);
    y[i] = (c < ca) ? a[m * ca + c] : b[m * cb + (c - ca)];
  }
}

__global__ void k_concat3(const float* __restrict__ a, const float* __restrict__ b,
                          const float* __restrict__ c_, int C, long long M, float* __restrict__ y) {
  int cy = 3 * C;
  long long total = M * cy;
  for (long long i = gtid(); i < total; i += gstr()) {
    long long m = i / cy; int c = (int)(i % cy);
    const float* s = (c < C) ? a : ((c < 2 * C) ? b : c_);
    int cc = (c < C) ? c : ((c < 2 * C) ? (c - C) : (c - 2 * C));
    y[i] = s[m * C + cc];
  }
}

// nm = [agg, agg^2, agg^4]  (both dense: agg width C, nm width 3C)
__global__ void k_moments3(const float* __restrict__ a, long long N, int C, float* __restrict__ y) {
  long long total = N * C;
  for (long long i = gtid(); i < total; i += gstr()) {
    long long n = i / C; int c = (int)(i % C);
    float v = a[i], v2 = v * v;
    float* row = y + n * (3LL * C);
    row[c] = v; row[C + c] = v2; row[2 * C + c] = v2 * v2;
  }
}

// ------------------------------ BN machinery -------------------------------

__global__ void k_colstats(const float* __restrict__ X, long long M, int K, int ld,
                           float* __restrict__ sum, float* __restrict__ sumsq) {
  __shared__ float ls[512], lq[512];
  for (int i = threadIdx.x; i < K; i += blockDim.x) { ls[i] = 0.f; lq[i] = 0.f; }
  __syncthreads();
  long long total = M * K;
  for (long long i = gtid(); i < total; i += gstr()) {
    long long m = i / K; int c = (int)(i % K);
    float v = X[m * ld + c];
    atomicAdd(&ls[c], v);
    atomicAdd(&lq[c], v * v);
  }
  __syncthreads();
  for (int i = threadIdx.x; i < K; i += blockDim.x) {
    atomicAdd(&sum[i], ls[i]);
    atomicAdd(&sumsq[i], lq[i]);
  }
}

__global__ void k_bn_finalize(const float* __restrict__ sum, const float* __restrict__ sumsq,
                              const float* __restrict__ gamma, const float* __restrict__ beta,
                              float invM, int K, float* __restrict__ scale, float* __restrict__ shift) {
  for (int i = threadIdx.x; i < K; i += blockDim.x) {
    float m = sum[i] * invM;
    float v = sumsq[i] * invM - m * m;
    v = v < 0.f ? 0.f : v;
    float s = gamma[i] * rsqrtf(v + BN_EPS);
    scale[i] = s;
    shift[i] = beta[i] - m * s;
  }
}

__global__ void k_bn_apply(float* __restrict__ Y, long long M, int K, int ld,
                           const float* __restrict__ scale, const float* __restrict__ shift, int leaky) {
  long long total = M * K;
  for (long long i = gtid(); i < total; i += gstr()) {
    long long m = i / K; int c = (int)(i % K);
    float v = Y[m * ld + c] * scale[c] + shift[c];
    if (leaky) v = (v > 0.f) ? v : 0.2f * v;
    Y[m * ld + c] = v;
  }
}

// ----------------------- weight packing (f32 -> padded bf16) ---------------

__global__ void k_pack_w(const float* __restrict__ W, int N, int K, int Np, int Kp,
                         __bf16* __restrict__ out) {
  long long total = (long long)Np * Kp;
  for (long long i = gtid(); i < total; i += gstr()) {
    int n = (int)(i / Kp), k = (int)(i % Kp);
    float v = (n < N && k < K) ? W[(long long)n * K + k] : 0.f;
    out[i] = (__bf16)v;
  }
}

// --------------------------- WMMA bf16 GEMM core ---------------------------
// Y[M, Np] = A[M, Kp] @ Wb[Np, Kp]^T + bias; padded lds, no guards in K loop.
// One wave: 16 rows x NT*32 cols (2*NT accumulators); NT is compile-time so
// the K loop is branch-free. A fragment loaded once per K-step, reused by
// 2*NT WMMA issues. Fused per-column sum/sumsq for training-mode BatchNorm
// (all rows are real: M % 16 == 0).

__device__ __forceinline__ bf16x16 load_a_frag(const float* __restrict__ p) {
  // lane's A slice: K = [0..15] contiguous relative to p
  f32x4 x0 = *(const f32x4*)(p);
  f32x4 x1 = *(const f32x4*)(p + 4);
  f32x4 x2 = *(const f32x4*)(p + 8);
  f32x4 x3 = *(const f32x4*)(p + 12);
  bf16x16 r;
#pragma unroll
  for (int j = 0; j < 4; ++j) {
    r[j]      = (__bf16)x0[j];
    r[4 + j]  = (__bf16)x1[j];
    r[8 + j]  = (__bf16)x2[j];
    r[12 + j] = (__bf16)x3[j];
  }
  return r;
}

template <int NT>
__global__ void __launch_bounds__(256)
k_gemm_bf16(const float* __restrict__ A, const __bf16* __restrict__ Wb, int Kp,
            const float* __restrict__ bias, int N,
            float* __restrict__ Y, int ldy, int Mt, int Nc,
            float* __restrict__ colsum, float* __restrict__ colsumsq) {
  const int lane = threadIdx.x & 31;
  const long long tile = (long long)blockIdx.x * (blockDim.x >> 5) + (threadIdx.x >> 5);
  if (tile >= (long long)Mt * Nc) return;            // wave-uniform
  const int tm = (int)(tile / Nc);
  const int tc = (int)(tile % Nc);
  const int cbase = tc * 128;

  const int l15 = lane & 15, lh = lane >> 4;
  const int aoff = lh * 16;                          // A k-sub-offset
  const int boff = lh * 16;                          // B k-sub-offset
  const float* __restrict__ Arow = A + (long long)(tm * 16 + l15) * Kp;

  const __bf16* __restrict__ Wp[NT];
#pragma unroll
  for (int j = 0; j < NT; ++j)
    Wp[j] = Wb + (long long)(cbase + j * 32 + l15) * Kp + boff;

  floatx8 acc[2 * NT] = {};
  for (int k0 = 0; k0 < Kp; k0 += 32) {
    __builtin_prefetch(Arow + k0 + 128, 0, 0);
    bf16x16 a = load_a_frag(Arow + k0 + aoff);
    bf16x16 b0[NT], b1[NT];
#pragma unroll
    for (int j = 0; j < NT; ++j) {                   // clause-able b128 loads
      b0[j] = *(const bf16x16*)(Wp[j] + k0);
      b1[j] = *(const bf16x16*)(Wp[j] + 16LL * Kp + k0);
    }
#pragma unroll
    for (int j = 0; j < NT; ++j) {                   // WMMA burst, A reused
      acc[2 * j]     = __builtin_amdgcn_wmma_f32_16x16x32_bf16(false, a, false, b0[j],
                                                               (short)0, acc[2 * j],     false, false);
      acc[2 * j + 1] = __builtin_amdgcn_wmma_f32_16x16x32_bf16(false, a, false, b1[j],
                                                               (short)0, acc[2 * j + 1], false, false);
    }
  }

  const int rb = tm * 16 + 8 * lh;                   // C/D layout: VGPR i -> row rb+i
#pragma unroll
  for (int j = 0; j < NT; ++j) {
    const int n0 = cbase + j * 32 + l15, n1 = n0 + 16;
    const bool ok0 = (n0 < N), ok1 = (n1 < N);
    const float bv0 = ok0 ? bias[n0] : 0.f;
    const float bv1 = ok1 ? bias[n1] : 0.f;
    float s0 = 0.f, q0 = 0.f, s1 = 0.f, q1 = 0.f;
#pragma unroll
    for (int i = 0; i < 8; ++i) {
      float d0 = acc[2 * j][i] + bv0;
      float d1 = acc[2 * j + 1][i] + bv1;
      Y[(long long)(rb + i) * ldy + n0] = d0;
      Y[(long long)(rb + i) * ldy + n1] = d1;
      s0 += d0; q0 += d0 * d0;
      s1 += d1; q1 += d1 * d1;
    }
    if (colsum != nullptr) {
      if (ok0) { atomicAdd(&colsum[n0], s0); atomicAdd(&colsumsq[n0], q0); }
      if (ok1) { atomicAdd(&colsum[n1], s1); atomicAdd(&colsumsq[n1], q1); }
    }
  }
}

// ------------------------------ graph kernels ------------------------------

// G[e] = [ e1[e], h[src[e]], h[dst[e]] ], zero pad to ldg
__global__ void k_gather_edge(const float* __restrict__ e1, int eld,
                              const float* __restrict__ h, int hd, int hld,
                              const int* __restrict__ src, const int* __restrict__ dst,
                              long long E, float* __restrict__ G, int ldg) {
  long long total = E * ldg;
  for (long long i = gtid(); i < total; i += gstr()) {
    long long ed = i / ldg; int c = (int)(i % ldg);
    float v = 0.f;
    if (c == 0)            v = e1[ed * eld];
    else if (c <= hd)      v = h[(long long)src[ed] * hld + (c - 1)];
    else if (c <= 2 * hd)  v = h[(long long)dst[ed] * hld + (c - 1 - hd)];
    G[i] = v;
  }
}

__global__ void k_scatter_add(const float* __restrict__ msg, const int* __restrict__ dst,
                              long long E, int C, float* __restrict__ agg) {
  long long total = E * C;
  for (long long i = gtid(); i < total; i += gstr()) {
    long long e = i / C; int c = (int)(i % C);
    atomicAdd(&agg[(long long)dst[e] * C + c], msg[i]);
  }
}

__global__ void k_scatter_max(const float* __restrict__ msg, const int* __restrict__ dst,
                              long long E, int C, float* __restrict__ agg) {
  long long total = E * C;
  for (long long i = gtid(); i < total; i += gstr()) {
    long long e = i / C; int c = (int)(i % C);
    atomicMaxF(&agg[(long long)dst[e] * C + c], msg[i]);
  }
}

__global__ void k_div_cnt(float* __restrict__ agg, const float* __restrict__ cnt, long long N, int C) {
  long long total = N * C;
  for (long long i = gtid(); i < total; i += gstr()) {
    float c = cnt[i / C];
    agg[i] /= (c > 1.f ? c : 1.f);
  }
}

__global__ void k_fix_max(float* __restrict__ a, long long n) {
  for (long long i = gtid(); i < n; i += gstr())
    if (!__builtin_isfinite(a[i])) a[i] = 0.f;
}

// ------------------------------ pooling / head -----------------------------

__global__ void k_seg_max(const float* __restrict__ g, int gld, const int* __restrict__ batch,
                          long long N, float* __restrict__ gmax) {
  for (long long i = gtid(); i < N; i += gstr()) atomicMaxF(&gmax[batch[i]], g[i * gld]);
}

__global__ void k_seg_expsum(const float* __restrict__ g, int gld, const int* __restrict__ batch,
                             const float* __restrict__ gmax, long long N,
                             float* __restrict__ ex, float* __restrict__ den) {
  for (long long i = gtid(); i < N; i += gstr()) {
    int b = batch[i];
    float v = expf(g[i * gld] - gmax[b]);
    ex[i] = v;
    atomicAdd(&den[b], v);
  }
}

__global__ void k_pool(const float* __restrict__ h, int C, const float* __restrict__ ex,
                       const float* __restrict__ den, const int* __restrict__ batch,
                       long long N, float* __restrict__ pooled) {
  long long total = N * C;
  for (long long i = gtid(); i < total; i += gstr()) {
    long long nn = i / C; int c = (int)(i % C);
    int b = batch[nn];
    float d = den[b];
    float w = (d > 0.f) ? (ex[nn] / d) : 0.f;
    atomicAdd(&pooled[(long long)b * C + c], w * h[i]);
  }
}

__global__ void k_final(const float* __restrict__ head, int hld, const float* __restrict__ alpha,
                        long long B, float* __restrict__ out) {
  long long total = B * LAT;
  for (long long i = gtid(); i < total; i += gstr()) {
    long long b = i / LAT; int c = (int)(i % LAT);
    out[i] = (c < LAT - 1) ? head[b * hld + c] : alpha[b];
  }
}

// ------------------------------- host side ---------------------------------

namespace {

struct Lay { const float *W, *b, *beta, *gamma; int cin, cout; };

struct ConvP {
  const float *bn_e_b, *bn_e_g, *bn_x_b, *bn_x_g, *f_bn_b, *f_bn_g;
  Lay f[3], g[4], mne[2], mnx[2];
  int xdim, outc, nmom;
};

inline int pad32(int x) { return (x + 31) & ~31; }

inline unsigned nblk(long long total) {
  long long b = (total + 255) / 256;
  if (b < 1) b = 1;
  if (b > 1048576) b = 1048576;
  return (unsigned)b;
}

// params leaves in jax pytree order: dict keys sorted, lists in order,
// per-layer dict sorted keys: W, b, be, g
int read_mlp_h(void* const* d_in, int cur, const int* dims, int nl, Lay* L) {
  for (int i = 0; i < nl; ++i) {
    L[i].W     = (const float*)d_in[cur++];
    L[i].b     = (const float*)d_in[cur++];
    L[i].beta  = (const float*)d_in[cur++];   // 'be'
    L[i].gamma = (const float*)d_in[cur++];   // 'g'
    L[i].cin = dims[i]; L[i].cout = dims[i + 1];
  }
  return cur;
}

// conv dict sorted keys: bn_e_b, bn_e_g, bn_x_b, bn_x_g, f, [f_bn_b, f_bn_g], g, mne, mnx
int read_conv(void* const* d_in, int cur, ConvP& C, int xdim, int outc, int nmom) {
  C.xdim = xdim; C.outc = outc; C.nmom = nmom;
  C.bn_e_b = (const float*)d_in[cur++]; C.bn_e_g = (const float*)d_in[cur++];
  C.bn_x_b = (const float*)d_in[cur++]; C.bn_x_g = (const float*)d_in[cur++];
  int fdims[4] = { nmom * outc, 128, 64, outc };
  cur = read_mlp_h(d_in, cur, fdims, 3, C.f);
  if (nmom > 1) { C.f_bn_b = (const float*)d_in[cur++]; C.f_bn_g = (const float*)d_in[cur++]; }
  else          { C.f_bn_b = nullptr; C.f_bn_g = nullptr; }
  int gdims[5] = { 1 + 2 * xdim, 128, 64, 32, outc };
  cur = read_mlp_h(d_in, cur, gdims, 4, C.g);
  int edims[3] = { 2, 2, 1 };
  cur = read_mlp_h(d_in, cur, edims, 2, C.mne);
  int xdims[3] = { 2 * xdim, 2 * xdim, xdim };
  cur = read_mlp_h(d_in, cur, xdims, 2, C.mnx);
  return cur;
}

struct WS {
  float *stats, *cnt;
  float *nT0, *nT1, *hnode, *x1, *x2, *x3, *xcat, *hbig, *agg, *nm, *gate, *ex;
  float *eG, *eT0, *eT1, *eA, *eB, *eOut;
  float *gmax, *den, *pooled, *hT0, *hT1, *hOut;
  __bf16* wbuf;
};

// Y gets ld = pad32(L.cout); A must have ld = pad32(L.cin) with zero pads.
void linear_bn(hipStream_t st, const float* A, long long M, const Lay& L,
               float* Y, WS& ws, bool leaky) {
  int Kp = pad32(L.cin), Np = pad32(L.cout);
  float *sum = ws.stats, *sumsq = ws.stats + 512, *scale = ws.stats + 1024, *shift = ws.stats + 1536;
  k_pack_w<<<nblk((long long)Np * Kp), 256, 0, st>>>(L.W, L.cout, L.cin, Np, Kp, ws.wbuf);
  k_fill<<<4, 256, 0, st>>>(sum, 1024, 0.f);
  int Mt = (int)(M / 16);                            // all M here are multiples of 16
  int Nc = (Np + 127) / 128;
  long long tiles = (long long)Mt * Nc;
  unsigned gb = (unsigned)((tiles + 7) / 8);
  int nt = (Np >= 128) ? 4 : (Np / 32);              // uniform across chunks for our Np set
  switch (nt) {
    case 1: k_gemm_bf16<1><<<gb, 256, 0, st>>>(A, ws.wbuf, Kp, L.b, L.cout, Y, Np, Mt, Nc, sum, sumsq); break;
    case 2: k_gemm_bf16<2><<<gb, 256, 0, st>>>(A, ws.wbuf, Kp, L.b, L.cout, Y, Np, Mt, Nc, sum, sumsq); break;
    default: k_gemm_bf16<4><<<gb, 256, 0, st>>>(A, ws.wbuf, Kp, L.b, L.cout, Y, Np, Mt, Nc, sum, sumsq); break;
  }
  k_bn_finalize<<<1, 512, 0, st>>>(sum, sumsq, L.gamma, L.beta, 1.f / (float)M, L.cout, scale, shift);
  k_bn_apply<<<nblk(M * L.cout), 256, 0, st>>>(Y, M, L.cout, Np, scale, shift, leaky ? 1 : 0);
}

void bn_standalone(hipStream_t st, float* X, long long M, int K, int ld,
                   const float* gamma, const float* beta, float* stats) {
  float *sum = stats, *sumsq = stats + 512, *scale = stats + 1024, *shift = stats + 1536;
  k_fill<<<4, 256, 0, st>>>(sum, 1024, 0.f);
  unsigned cb = nblk(M * K); if (cb > 2048) cb = 2048;
  k_colstats<<<cb, 256, 0, st>>>(X, M, K, ld, sum, sumsq);
  k_bn_finalize<<<1, 512, 0, st>>>(sum, sumsq, gamma, beta, 1.f / (float)M, K, scale, shift);
  k_bn_apply<<<nblk(M * K), 256, 0, st>>>(X, M, K, ld, scale, shift, 0);
}

const float* run_mlp(hipStream_t st, const float* in, long long M,
                     const Lay* L, int nl, float* t0, float* t1, float* out, WS& ws) {
  const float* cur = in;
  for (int i = 0; i < nl; ++i) {
    float* o = (i == nl - 1) ? out : ((i & 1) ? t1 : t0);
    linear_bn(st, cur, M, L[i], o, ws, i < nl - 1);
    cur = o;
  }
  return cur;
}

void run_conv(hipStream_t st, const ConvP& C, const float* x, const float* eattr,
              const int* src, const int* dst, int aggr_max, float* out, WS& ws) {
  // node branch: [x, x^2] -> BN -> mnx           (x dense width xdim)
  int ld2x = pad32(2 * C.xdim);
  k_concat_sq<<<nblk(NNODES * ld2x), 256, 0, st>>>(x, NNODES, C.xdim, ld2x, ws.nT0);
  bn_standalone(st, ws.nT0, NNODES, 2 * C.xdim, ld2x, C.bn_x_g, C.bn_x_b, ws.stats);
  run_mlp(st, ws.nT0, NNODES, C.mnx, 2, ws.nT1, ws.nT1, ws.hnode, ws);
  // edge branch: [e, e^2] -> BN -> mne
  k_concat_sq<<<nblk(NEDGES * 32), 256, 0, st>>>(eattr, NEDGES, 1, 32, ws.eA);
  bn_standalone(st, ws.eA, NEDGES, 2, 32, C.bn_e_g, C.bn_e_b, ws.stats);
  run_mlp(st, ws.eA, NEDGES, C.mne, 2, ws.eB, ws.eB, ws.eOut, ws);
  // per-edge message MLP g on [e, h_src, h_dst]
  int kg = 1 + 2 * C.xdim, ldg = pad32(kg), hld = pad32(C.xdim);
  k_gather_edge<<<nblk(NEDGES * ldg), 256, 0, st>>>(ws.eOut, 32, ws.hnode, C.xdim, hld,
                                                    src, dst, NEDGES, ws.eG, ldg);
  run_mlp(st, ws.eG, NEDGES, C.g, 4, ws.eT0, ws.eT1, ws.eT1, ws);
  const float* msg = ws.eT1;                                    // [E, outc] dense (64)
  // scatter aggregate
  long long aggn = NNODES * C.outc;
  if (!aggr_max) {
    k_fill<<<nblk(aggn), 256, 0, st>>>(ws.agg, aggn, 0.f);
    k_scatter_add<<<nblk(NEDGES * C.outc), 256, 0, st>>>(msg, dst, NEDGES, C.outc, ws.agg);
    k_div_cnt<<<nblk(aggn), 256, 0, st>>>(ws.agg, ws.cnt, NNODES, C.outc);
  } else {
    k_fill<<<nblk(aggn), 256, 0, st>>>(ws.agg, aggn, -__builtin_huge_valf());
    k_scatter_max<<<nblk(NEDGES * C.outc), 256, 0, st>>>(msg, dst, NEDGES, C.outc, ws.agg);
    k_fix_max<<<nblk(aggn), 256, 0, st>>>(ws.agg, aggn);
  }
  // moments -> (f_bn) -> f
  const float* fin = ws.agg;
  if (C.nmom == 3) {
    k_moments3<<<nblk(aggn), 256, 0, st>>>(ws.agg, NNODES, C.outc, ws.nm);
    bn_standalone(st, ws.nm, NNODES, 3 * C.outc, 3 * C.outc, C.f_bn_g, C.f_bn_b, ws.stats);
    fin = ws.nm;
  }
  run_mlp(st, fin, NNODES, C.f, 3, ws.nT0, ws.nT1, out, ws);
}

} // namespace

extern "C" void kernel_launch(void* const* d_in, const int* in_sizes, int n_in,
                              void* d_out, int out_size, void* d_ws, size_t ws_size,
                              hipStream_t stream) {
  (void)in_sizes; (void)n_in; (void)out_size; (void)ws_size;
  const float* x     = (const float*)d_in[0];
  const float* eattr = (const float*)d_in[1];
  const float* alpha = (const float*)d_in[2];

  int cur = 3;
  ConvP c1, c2, c3;
  cur = read_conv(d_in, cur, c1, 1,       NCH, 3);
  cur = read_conv(d_in, cur, c2, NCH,     NCH, 1);
  cur = read_conv(d_in, cur, c3, 2 * NCH, NCH, 3);
  Lay gateL[3]; { int d[4] = { 3 * NCH, NCH, NCH / 2, 1 };           cur = read_mlp_h(d_in, cur, d, 3, gateL); }
  Lay headL[3]; { int d[4] = { 3 * NCH, 2 * NCH, 2 * LAT, LAT - 1 }; cur = read_mlp_h(d_in, cur, d, 3, headL); }
  const int* ei    = (const int*)d_in[cur++];
  const int* batch = (const int*)d_in[cur++];
  const int* src = ei;
  const int* dst = ei + NEDGES;

  // deterministic workspace carve
  char* p = (char*)d_ws;
  auto carve = [&](long long elems) -> float* {
    float* r = (float*)p;
    p += (size_t)(((elems * 4 + 255) / 256) * 256);
    return r;
  };
  WS ws;
  ws.stats  = carve(2048);
  ws.cnt    = carve(NNODES);
  ws.nT0    = carve(NNODES * 256);
  ws.nT1    = carve(NNODES * 256);
  ws.hnode  = carve(NNODES * 128);
  ws.x1     = carve(NNODES * NCH);
  ws.x2     = carve(NNODES * NCH);
  ws.x3     = carve(NNODES * NCH);
  ws.xcat   = carve(NNODES * 2 * NCH);
  ws.hbig   = carve(NNODES * 3 * NCH);
  ws.agg    = carve(NNODES * NCH);
  ws.nm     = carve(NNODES * 3 * NCH);
  ws.gate   = carve(NNODES * 32);
  ws.ex     = carve(NNODES);
  ws.eG     = carve(NEDGES * 288);
  ws.eT0    = carve(NEDGES * 128);
  ws.eT1    = carve(NEDGES * 64);
  ws.eA     = carve(NEDGES * 32);
  ws.eB     = carve(NEDGES * 32);
  ws.eOut   = carve(NEDGES * 32);
  ws.gmax   = carve(NGRAPH);
  ws.den    = carve(NGRAPH);
  ws.pooled = carve(NGRAPH * 3 * NCH);
  ws.hT0    = carve(NGRAPH * 128);
  ws.hT1    = carve(NGRAPH * 256);
  ws.hOut   = carve(NGRAPH * 128);
  ws.wbuf   = (__bf16*)carve(49152);    // 96K bf16 >= max Np*Kp (256*256)

  // in-degree counts (shared by all mean aggregations)
  k_fill<<<nblk(NNODES), 256, 0, stream>>>(ws.cnt, NNODES, 0.f);
  k_count_edges<<<nblk(NEDGES), 256, 0, stream>>>(dst, NEDGES, ws.cnt);

  // three JumpsConv layers
  run_conv(stream, c1, x,       eattr, src, dst, /*max=*/0, ws.x1, ws);
  run_conv(stream, c2, ws.x1,   eattr, src, dst, /*max=*/1, ws.x2, ws);
  k_concat2<<<nblk(NNODES * 2 * NCH), 256, 0, stream>>>(ws.x1, NCH, ws.x2, NCH, NNODES, ws.xcat);
  run_conv(stream, c3, ws.xcat, eattr, src, dst, /*max=*/0, ws.x3, ws);
  k_concat3<<<nblk(NNODES * 3 * NCH), 256, 0, stream>>>(ws.x3, ws.x1, ws.x2, NCH, NNODES, ws.hbig);

  // GlobalAttention: gate MLP -> segment softmax -> weighted segment sum
  run_mlp(stream, ws.hbig, NNODES, gateL, 3, ws.nT0, ws.nT1, ws.gate, ws);
  k_fill<<<nblk(NGRAPH), 256, 0, stream>>>(ws.gmax, NGRAPH, -__builtin_huge_valf());
  k_seg_max<<<nblk(NNODES), 256, 0, stream>>>(ws.gate, 32, batch, NNODES, ws.gmax);
  k_fill<<<nblk(NGRAPH), 256, 0, stream>>>(ws.den, NGRAPH, 0.f);
  k_seg_expsum<<<nblk(NNODES), 256, 0, stream>>>(ws.gate, 32, batch, ws.gmax, NNODES, ws.ex, ws.den);
  k_fill<<<nblk(NGRAPH * 3 * NCH), 256, 0, stream>>>(ws.pooled, NGRAPH * 3 * NCH, 0.f);
  k_pool<<<nblk(NNODES * 3 * NCH), 256, 0, stream>>>(ws.hbig, 3 * NCH, ws.ex, ws.den, batch, NNODES, ws.pooled);

  // head MLP on pooled graph features + alpha concat
  run_mlp(stream, ws.pooled, NGRAPH, headL, 3, ws.hT0, ws.hT1, ws.hOut, ws);
  k_final<<<nblk(NGRAPH * LAT), 256, 0, stream>>>(ws.hOut, 128, alpha, NGRAPH, (float*)d_out);
}
